// Band_selection_attn_83442624627151
// MI455X (gfx1250) — matrix operand order repaired and checked by reading, not compiled, not run on hardware
//
#include <hip/hip_runtime.h>

// ---------------- problem constants ----------------
#define BAND 25
#define E    32
#define H    8
#define Dh   4
#define NB   16384
#define LSH  40      // f16 LDS row stride (halves): 80B rows -> 16B-aligned k-runs
#define LSF  36      // f32 LDS row stride (floats): 144B rows -> 16B-aligned
#define WAVES 2      // batches (waves) per workgroup

typedef __attribute__((ext_vector_type(16))) _Float16 v16h;
typedef __attribute__((ext_vector_type(8)))  _Float16 v8h;
typedef __attribute__((ext_vector_type(4)))  _Float16 v4h;
typedef __attribute__((ext_vector_type(8)))  float    v8f;
typedef __attribute__((ext_vector_type(4)))  float    v4f;

__device__ __forceinline__ v8f wmma_f16(v16h a, v16h b, v8f c) {
  // D = A(16x32 f16) * B(32x16 f16) + C(16x16 f32)
  return __builtin_amdgcn_wmma_f32_16x16x32_f16(
      /*neg_a=*/false, a, /*neg_b=*/false, b,
      /*c_mod=*/(short)0, c, /*reuse_a=*/false, /*reuse_b=*/false);
}

__device__ __forceinline__ v16h cat8(v8h lo, v8h hi) {
  return __builtin_shufflevector(lo, hi, 0, 1, 2, 3, 4, 5, 6, 7,
                                 8, 9, 10, 11, 12, 13, 14, 15);
}

// ---- A fragment (16x32 f16): lane m=lane&15, kh=lane>>4 holds
//      K = {16kh..16kh+15} in a[0..15];
//      a contiguous 32B run of a row-major LDS row -> 2x ds_load_b128.
__device__ __forceinline__ v16h frag_a_f16(const _Float16* base, int row0, int lane) {
  const int m = lane & 15, kh = lane >> 4;
  const _Float16* r = base + (row0 + m) * LSH;
  v8h lo = *(const v8h*)(r + 16 * kh);
  v8h hi = *(const v8h*)(r + 16 * kh + 8);
  return cat8(lo, hi);
}

// Same A fragment sourced from an f32 LDS matrix: 4x ds_load_b128 + cvt.
__device__ __forceinline__ v16h frag_a_f32(const float* base, int row0, int lane) {
  const int m = lane & 15, kh = lane >> 4;
  const float* r = base + (row0 + m) * LSF;
  v4f f0 = *(const v4f*)(r + 16 * kh);
  v4f f1 = *(const v4f*)(r + 16 * kh + 4);
  v4f f2 = *(const v4f*)(r + 16 * kh + 8);
  v4f f3 = *(const v4f*)(r + 16 * kh + 12);
  v8h lo = __builtin_shufflevector(__builtin_convertvector(f0, v4h),
                                   __builtin_convertvector(f1, v4h),
                                   0, 1, 2, 3, 4, 5, 6, 7);
  v8h hi = __builtin_shufflevector(__builtin_convertvector(f2, v4h),
                                   __builtin_convertvector(f3, v4h),
                                   0, 1, 2, 3, 4, 5, 6, 7);
  return cat8(lo, hi);
}

// ---- B fragment where B = M^T (B[k][n] = M[n][k]): lane n=lane&15, kh=lane>>4
//      needs K = {16kh..16kh+15} of row (col0+n): one contiguous 32B run.
__device__ __forceinline__ v16h frag_b_tr(const _Float16* base, int col0, int lane) {
  const int n = lane & 15, kh = lane >> 4;
  const _Float16* r = base + (col0 + n) * LSH + 16 * kh;
  v8h lo = *(const v8h*)(r);
  v8h hi = *(const v8h*)(r + 8);
  return cat8(lo, hi);
}

// ---- B fragment from pre-packed per-lane storage (weights): 2x ds_load_b128.
__device__ __forceinline__ v16h frag_b_packed(const _Float16* p) {
  v8h lo = *(const v8h*)(p);
  v8h hi = *(const v8h*)(p + 8);
  return cat8(lo, hi);
}

// =====================================================================
// Kernel 1: per-batch attention + raw correlation matrix + global max
// =====================================================================
__global__ __launch_bounds__(WAVES * 32)
void bandsel_attn_main(const float* __restrict__ x_template,
                       const float* __restrict__ x_search,
                       const float* __restrict__ in_proj_w,   // (96,32)
                       const float* __restrict__ in_proj_b,   // (96,)
                       const float* __restrict__ out_proj_w,  // (32,32)
                       const float* __restrict__ out_proj_b,  // (32,)
                       float* __restrict__ cm_out,            // (NB,25,25) raw
                       unsigned int* __restrict__ gmax) {
  // Weights pre-packed in exact B-fragment order: [mat][colTile][lane][16]
  __shared__ _Float16 sWB[4][2][32][16];
  __shared__ float    sBias[4][32];
  __shared__ _Float16 sM0[WAVES][32 * LSH];   // xs^T (f16, padded) -> later W0
  __shared__ _Float16 sM1[WAVES][32 * LSH];   // xt^T (f16, padded)
  __shared__ float    sQ[WAVES][32 * LSF];
  __shared__ float    sK[WAVES][32 * LSF];
  __shared__ float    sV[WAVES][32 * LSF];
  __shared__ float    sS[WAVES][32 * LSF];    // per-head scores / probs
  __shared__ float    sO[WAVES][32 * LSF];    // attention output (pre out-proj)

  const int tid  = threadIdx.x;
  const int lane = tid & 31;
  const int w    = tid >> 5;
  const int b    = blockIdx.x * WAVES + w;
  const int nl   = lane & 15, kh = lane >> 4;

  // ---- stage weights once per block, directly in B-fragment order ----
  // element q of lane L, tile cb, matrix `mat` is W[n][k] with
  //   n = cb*16 + (L&15),  k = 16*(L>>4) + q
  for (int idx = tid; idx < 4 * 2 * 32 * 16; idx += WAVES * 32) {
    const int mat   = idx >> 10;
    const int rem   = idx & 1023;
    const int cb    = rem >> 9;
    const int lane2 = (rem >> 4) & 31;
    const int q     = rem & 15;
    const int n     = cb * 16 + (lane2 & 15);
    const int k     = 16 * (lane2 >> 4) + q;
    const float v = (mat < 3) ? in_proj_w[(mat * 32 + n) * 32 + k]
                              : out_proj_w[n * 32 + k];
    sWB[mat][cb][lane2][q] = (_Float16)v;
  }
  for (int idx = tid; idx < 96; idx += WAVES * 32) sBias[idx >> 5][idx & 31] = in_proj_b[idx];
  for (int idx = tid; idx < 32; idx += WAVES * 32) sBias[3][idx] = out_proj_b[idx];

  _Float16* M0 = sM0[w];
  _Float16* M1 = sM1[w];
  float *Q = sQ[w], *K = sK[w], *V = sV[w], *S = sS[w], *O = sO[w];

  // zero-init padded tiles + O accumulator
  for (int i = lane; i < 32 * LSH; i += 32) { M0[i] = (_Float16)0.f; M1[i] = (_Float16)0.f; }
  for (int i = lane; i < 32 * LSF; i += 32) O[i] = 0.f;
  __syncthreads();

  // ---- load xs^T / xt^T : input (E=32, BAND=25) row-major -> [l][e] f16 ----
  const float* xs = x_search   + (size_t)b * (E * BAND);
  const float* xt = x_template + (size_t)b * (E * BAND);
  for (int idx = lane; idx < E * BAND; idx += 32) {
    const int e = idx / BAND, l = idx % BAND;
    M0[l * LSH + e] = (_Float16)xs[idx];
    M1[l * LSH + e] = (_Float16)xt[idx];
  }
  __syncthreads();

  // ---- Q/K/V projections (WMMA): Out = X @ W^T + bias ----
  {
    const _Float16* Xs[3] = {M0, M1, M1};
    float* Outs[3] = {Q, K, V};
#pragma unroll
    for (int mat = 0; mat < 3; ++mat) {
#pragma unroll
      for (int r = 0; r < 2; ++r) {
        v16h a = frag_a_f16(Xs[mat], r * 16, lane);
#pragma unroll
        for (int cb = 0; cb < 2; ++cb) {
          v16h bb = frag_b_packed(&sWB[mat][cb][lane][0]);
          v8f c = {};
          c = wmma_f16(a, bb, c);
          const int n = cb * 16 + nl;
          const float bv = sBias[mat][n];
          float* outp = Outs[mat];
#pragma unroll
          for (int p = 0; p < 8; ++p) {
            const int m = r * 16 + p + 8 * kh;  // C layout: lanes0-15 M=p, 16-31 M=p+8
            outp[m * LSF + n] = c[p] + bv;
          }
        }
      }
    }
  }
  __syncthreads();

  // ---- per-head scores -> softmax -> P @ V_h  (D=4: VALU, float4 slices) ----
  for (int h = 0; h < H; ++h) {
    const int e0 = h * Dh;
    for (int idx = lane; idx < BAND * BAND; idx += 32) {
      const int i = idx / BAND, j = idx % BAND;
      v4f qv = *(const v4f*)(Q + i * LSF + e0);
      v4f kv = *(const v4f*)(K + j * LSF + e0);
      v4f t = qv * kv;
      S[i * LSF + j] = (t.x + t.y + t.z + t.w) * 0.5f;   // 1/sqrt(D), D=4
    }
    __syncthreads();
    if (lane < BAND) {
      float mx = -1e30f;
      for (int j = 0; j < BAND; ++j) mx = fmaxf(mx, S[lane * LSF + j]);
      float sum = 0.f;
      for (int j = 0; j < BAND; ++j) {
        const float ev = __expf(S[lane * LSF + j] - mx);
        S[lane * LSF + j] = ev;
        sum += ev;
      }
      const float inv = 1.f / sum;
      for (int j = 0; j < BAND; ++j) S[lane * LSF + j] *= inv;
    }
    __syncthreads();
    if (lane < BAND) {
      v4f acc = {0.f, 0.f, 0.f, 0.f};
      for (int j = 0; j < BAND; ++j) {
        const float p = S[lane * LSF + j];
        v4f vv = *(const v4f*)(V + j * LSF + e0);
        acc += vv * p;
      }
      *(v4f*)(O + lane * LSF + e0) = acc;    // disjoint columns per head
    }
    __syncthreads();
  }

  // ---- out-proj + bias + ReLU  ->  W0 (f16) into M0, pad rows forced to 0 ----
#pragma unroll
  for (int r = 0; r < 2; ++r) {
    v16h a = frag_a_f32(O, r * 16, lane);
#pragma unroll
    for (int cb = 0; cb < 2; ++cb) {
      v16h bb = frag_b_packed(&sWB[3][cb][lane][0]);
      v8f c = {};
      c = wmma_f16(a, bb, c);
      const int n = cb * 16 + nl;
      const float bv = sBias[3][n];
#pragma unroll
      for (int p = 0; p < 8; ++p) {
        const int m = r * 16 + p + 8 * kh;
        float v = c[p] + bv;
        v = v > 0.f ? v : 0.f;
        M0[m * LSH + n] = (_Float16)((m < BAND) ? v : 0.f);
      }
    }
  }
  __syncthreads();

  // ---- cm = W0 @ W0^T  (WMMA, B read transposed from LDS) + global max ----
  float* cmb = cm_out + (size_t)b * (BAND * BAND);
  float lmax = 0.f;
#pragma unroll
  for (int r = 0; r < 2; ++r) {
    v16h a = frag_a_f16(M0, r * 16, lane);
#pragma unroll
    for (int cb = 0; cb < 2; ++cb) {
      v16h bb = frag_b_tr(M0, cb * 16, lane);
      v8f c = {};
      c = wmma_f16(a, bb, c);
      const int n = cb * 16 + nl;
      if (n < BAND) {
#pragma unroll
        for (int p = 0; p < 8; ++p) {
          const int m = r * 16 + p + 8 * kh;
          if (m < BAND) {
            const float v = c[p];
            cmb[m * BAND + n] = v;
            lmax = fmaxf(lmax, v);          // cm >= 0 (relu . relu)
          }
        }
      }
    }
  }
  // wave-wide max reduction, one atomic per wave
  for (int off = 16; off > 0; off >>= 1)
    lmax = fmaxf(lmax, __shfl_xor(lmax, off, 32));
  if (lane == 0) atomicMax(gmax, __float_as_uint(lmax));
}

// =====================================================================
// Kernel 2: normalize, zero diagonal, row means, stable descending sort
// =====================================================================
__global__ __launch_bounds__(32)
void bandsel_attn_finish(float* __restrict__ vals,
                         float* __restrict__ order,
                         float* __restrict__ cm,
                         const unsigned int* __restrict__ gmax) {
  __shared__ float ybuf[BAND];
  const int b = blockIdx.x;
  const int t = threadIdx.x;
  const float inv = 1.f / __uint_as_float(*gmax);
  float* cmb = cm + (size_t)b * (BAND * BAND);

  if (t < BAND) {
    float sum = 0.f;
    for (int j = 0; j < BAND; ++j) {
      float v = cmb[t * BAND + j] * inv;
      if (j == t) v = 0.f;
      cmb[t * BAND + j] = v;
      sum += v;
    }
    ybuf[t] = sum * (1.f / BAND);
  }
  __syncthreads();
  if (t < BAND) {
    const float y = ybuf[t];
    int rank = 0;
    for (int j = 0; j < BAND; ++j) {
      const float yj = ybuf[j];
      rank += (yj > y) || (yj == y && j < t);   // stable argsort(-y)
    }
    vals[(size_t)b * BAND + rank]  = y;
    order[(size_t)b * BAND + rank] = (float)t;
  }
}

// =====================================================================
extern "C" void kernel_launch(void* const* d_in, const int* in_sizes, int n_in,
                              void* d_out, int out_size, void* d_ws, size_t ws_size,
                              hipStream_t stream) {
  const float* x_template = (const float*)d_in[0];
  const float* x_search   = (const float*)d_in[1];
  const float* in_proj_w  = (const float*)d_in[2];
  const float* in_proj_b  = (const float*)d_in[3];
  const float* out_proj_w = (const float*)d_in[4];
  const float* out_proj_b = (const float*)d_in[5];

  float* out   = (float*)d_out;
  float* vals  = out;                              // (NB, 25)
  float* order = out + (size_t)NB * BAND;          // (NB, 25) as float
  float* cm    = out + (size_t)2 * NB * BAND;      // (NB, 25, 25)
  unsigned int* gmax = (unsigned int*)d_ws;

  hipMemsetAsync(d_ws, 0, sizeof(unsigned int), stream);  // capture-safe

  bandsel_attn_main<<<NB / WAVES, WAVES * 32, 0, stream>>>(
      x_template, x_search, in_proj_w, in_proj_b, out_proj_w, out_proj_b,
      cm, gmax);

  bandsel_attn_finish<<<NB, 32, 0, stream>>>(vals, order, cm, gmax);
}
